// SelfMatcher_25744033973070
// MI455X (gfx1250) — matrix operand order repaired and checked by reading, not compiled, not run on hardware
//
#include <hip/hip_runtime.h>
#include <hip/hip_bf16.h>
#include <math.h>

// SelfMatcher for MI455X (gfx1250, wave32, WMMA + TDM).
// B=48, L=512, D=H=384.
//
//   1) wmma_gemm: Wpv = v @ Wp_seq           [24576,384]
//   2) wmma_gemm: Wpi = v @ Wp_cur           [24576,384]
//   3) attention_kernel (fused tanh-score/softmax/context), block per (b,i)
//   4) wmma_gemm: GI = [v|C] @ W_ih^T + b_ih [24576,1152]
//   5) transpose_whh, then gru_kernel: 48 independent recurrences.
//
// GEMM: 128x64 tile per workgroup (8 waves), B tile staged in LDS via the
// Tensor Data Mover (tensor_load_to_lds + s_wait_tensorcnt) when available.

namespace sm {
constexpr int B = 48, L = 512, D = 384, H = 384;
constexpr int M = B * L;          // 24576 rows
constexpr int G3 = 3 * H;         // 1152
}

typedef __attribute__((ext_vector_type(16))) _Float16 v16h;
typedef __attribute__((ext_vector_type(8)))  float    v8f;
typedef __attribute__((ext_vector_type(4)))  unsigned u32x4;
typedef __attribute__((ext_vector_type(4)))  int      i32x4;
typedef __attribute__((ext_vector_type(8)))  int      i32x8;

#if __has_builtin(__builtin_amdgcn_tensor_load_to_lds)
#define SM_HAS_TDM 1
#endif

#ifdef SM_HAS_TDM
// Issue a 2-D TDM tile load Global->LDS.  D# packed per CDNA5 ISA §8.3-8.5:
//   group0: count=1, lds_addr, 57-bit global tile address, type=2 (image)
//   group1: data_size=4B, tensor_dim0/1, tile_dim0/1, tensor_dim0_stride
// tile is tile_d1 rows of tile_d0 contiguous f32, row pitch = stride_d0 elems.
__device__ inline void tdm_load_tile_f32(unsigned lds_off, const float* gptr,
                                         unsigned tensor_d0, unsigned tensor_d1,
                                         unsigned tile_d0, unsigned tile_d1,
                                         unsigned stride_d0) {
  const unsigned long long ga = (unsigned long long)(uintptr_t)gptr;
  u32x4 g0;
  g0[0] = 1u;                                        // count=1, user D#
  g0[1] = lds_off;                                   // LDS byte address
  g0[2] = (unsigned)(ga & 0xffffffffu);              // global_addr[31:0]
  g0[3] = (unsigned)((ga >> 32) & 0x01ffffffu)       // global_addr[56:32]
        | (2u << 30);                                // type=2
  i32x8 g1;
  g1[0] = (int)(2u << 16);                           // data_size=2 -> 4 bytes
  g1[1] = (int)((tensor_d0 & 0xffffu) << 16);        // tensor_dim0[15:0]
  g1[2] = (int)(((tensor_d0 >> 16) & 0xffffu)        // tensor_dim0[31:16]
        | ((tensor_d1 & 0xffffu) << 16));            // tensor_dim1[15:0]
  g1[3] = (int)(((tensor_d1 >> 16) & 0xffffu)        // tensor_dim1[31:16]
        | ((tile_d0 & 0xffffu) << 16));              // tile_dim0
  g1[4] = (int)(tile_d1 & 0xffffu);                  // tile_dim1 (tile_dim2=0)
  g1[5] = (int)stride_d0;                            // tensor_dim0_stride[31:0]
  g1[6] = 0;                                         // stride hi / dim1_stride
  g1[7] = 0;
  i32x4 gz = {0, 0, 0, 0};
#if __clang_major__ >= 23
  i32x8 gz8 = {0, 0, 0, 0, 0, 0, 0, 0};
  __builtin_amdgcn_tensor_load_to_lds(g0, g1, gz, gz, gz8, 0);
#else
  __builtin_amdgcn_tensor_load_to_lds(g0, g1, gz, gz, 0);
#endif
}
#endif

// ---------------------------------------------------------------------------
// WMMA GEMM: Y[M,N] = A[M,K] @ W (+ bias)
//   WT=0 : W is [K,N] row-major.   WT=1 : W is [N,K] row-major (i.e. @ W^T)
//   CONCAT: A = [X | X2] along K, both row stride K1 (K = 2*K1)
// Block (32,8): 8 waves; wave w -> rows [blockRow + 16w, +16), cols 64.
// B tile (32K x 64N f32) staged in LDS once per K-step, shared by all waves.
// Each wave: 1 A-fragment -> 4 WMMAs per K-step.
// Grid: (N/64, M/128).
// ---------------------------------------------------------------------------
template <bool WT, bool CONCAT, bool BIAS>
__global__ __launch_bounds__(256)
void wmma_gemm(const float* __restrict__ X,
               const float* __restrict__ X2,
               const float* __restrict__ W,
               const float* __restrict__ bias,
               float* __restrict__ Y,
               int K, int N, int K1) {
  const int lane = threadIdx.x;                 // 0..31
  const int wave = threadIdx.y;                 // 0..7
  const int tid  = wave * 32 + lane;
  const int col0 = blockIdx.x * 64;
  const int row0 = (blockIdx.y * 8 + wave) * 16;
  const int mrow = lane & 15;                   // A: M idx, B/D: N idx
  const int half = lane >> 4;

  // LDS B tile: !WT -> [k(32)][n(64)] ; WT -> [n(64)][k(32)]  (2048 f32, 8KB)
  __shared__ float Blds[32 * 64];

  v8f acc0 = {}, acc1 = {}, acc2 = {}, acc3 = {};

  for (int k0 = 0; k0 < K; k0 += 32) {
    __syncthreads();                            // previous tile consumed
#ifdef SM_HAS_TDM
    if (wave == 0) {                            // one TDM op per workgroup
      const unsigned lds_off = (unsigned)(uintptr_t)&Blds[0];
      if constexpr (WT)                         // tile: 64 rows(n) x 32(k)
        tdm_load_tile_f32(lds_off, W + (size_t)col0 * K + k0,
                          (unsigned)K, (unsigned)N, 32u, 64u, (unsigned)K);
      else                                      // tile: 32 rows(k) x 64(n)
        tdm_load_tile_f32(lds_off, W + (size_t)k0 * N + col0,
                          (unsigned)N, (unsigned)K, 64u, 32u, (unsigned)N);
      __builtin_amdgcn_s_wait_tensorcnt(0);
    }
#else
    for (int idx = tid; idx < 32 * 64; idx += 256) {
      if constexpr (WT) {                       // Blds[n][k]
        const int n = idx >> 5, k = idx & 31;
        Blds[idx] = W[(size_t)(col0 + n) * K + (k0 + k)];
      } else {                                  // Blds[k][n]
        const int k = idx >> 6, n = idx & 63;
        Blds[idx] = W[(size_t)(k0 + k) * N + (col0 + n)];
      }
    }
#endif
    __syncthreads();

    // ---- A fragment: 16x32 f16 (ISA 7.12.2 layout) ----
    v16h a;
#pragma unroll
    for (int e = 0; e < 16; ++e) {
      const int kk = k0 + ((e < 8) ? (half * 8 + e) : (16 + half * 8 + (e - 8)));
      float x;
      if constexpr (CONCAT) {
        x = (kk < K1) ? X[(size_t)(row0 + mrow) * K1 + kk]
                      : X2[(size_t)(row0 + mrow) * K1 + (kk - K1)];
      } else {
        x = X[(size_t)(row0 + mrow) * K + kk];
      }
      a[e] = (_Float16)x;
    }

    // ---- 4 B fragments from LDS, 4 WMMAs ----
#pragma unroll
    for (int c = 0; c < 4; ++c) {
      v16h b;
#pragma unroll
      for (int e = 0; e < 16; ++e) {
        const int kk = (e < 8) ? (half * 8 + e) : (16 + half * 8 + (e - 8));           // local k 0..31
        const int nn = c * 16 + mrow;           // local n 0..63
        b[e] = (_Float16)(WT ? Blds[nn * 32 + kk] : Blds[kk * 64 + nn]);
      }
      v8f* accp = (c == 0) ? &acc0 : (c == 1) ? &acc1 : (c == 2) ? &acc2 : &acc3;
      *accp = __builtin_amdgcn_wmma_f32_16x16x32_f16(
          false, a, false, b, (short)0, *accp, false, false);
    }
  }

  // ---- D store: VGPR r -> M = row0 + half*8 + r ----
#pragma unroll
  for (int c = 0; c < 4; ++c) {
    const v8f acc = (c == 0) ? acc0 : (c == 1) ? acc1 : (c == 2) ? acc2 : acc3;
    const int nn = col0 + c * 16 + mrow;
    const float bv = BIAS ? bias[nn] : 0.0f;
#pragma unroll
    for (int r = 0; r < 8; ++r) {
      const int mm = row0 + half * 8 + r;
      Y[(size_t)mm * N + nn] = acc[r] + bv;
    }
  }
}

// ---------------------------------------------------------------------------
// Fused attention: one block per (b,i).  384 threads = 12 waves.
// ---------------------------------------------------------------------------
__global__ __launch_bounds__(384)
void attention_kernel(const float* __restrict__ Wpv,
                      const float* __restrict__ Wpi,
                      const float* __restrict__ Vw,
                      const float* __restrict__ v,
                      const unsigned char* __restrict__ mask,
                      float* __restrict__ Cout) {
  using namespace sm;
  constexpr int BDIM = 384, NW = BDIM / 32;
  const int bi = blockIdx.x;            // b*L + i
  const int b  = bi / L;
  const int t = threadIdx.x, lane = t & 31, wave = t >> 5;

  __shared__ float wpi_s[H];
  __shared__ float vs_s[H];
  __shared__ float s_s[L];
  __shared__ float red_s[NW];

  wpi_s[t] = Wpi[(size_t)bi * H + t];   // BDIM == H
  vs_s[t]  = Vw[t];
  __syncthreads();

  // scores: wave w -> rows l = w, w+NW, ... (coalesced over h within wave)
  for (int l = wave; l < L; l += NW) {
    const float* wrow = Wpv + ((size_t)b * L + l) * H;
    __builtin_prefetch(wrow + ((l + NW) < L ? (size_t)NW * H : 0), 0, 0);
    float acc = 0.0f;
    for (int h = lane; h < H; h += 32)
      acc += vs_s[h] * tanhf(wrow[h] + wpi_s[h]);
#pragma unroll
    for (int off = 16; off > 0; off >>= 1) acc += __shfl_xor(acc, off, 32);
    if (lane == 0) {
      const float neg = mask[(size_t)b * L + l] ? 0.0f : -1e30f;
      s_s[l] = acc + neg;
    }
  }
  __syncthreads();

  // softmax over L=512
  float lm = -3.0e38f;
  for (int l = t; l < L; l += BDIM) lm = fmaxf(lm, s_s[l]);
#pragma unroll
  for (int off = 16; off > 0; off >>= 1) lm = fmaxf(lm, __shfl_xor(lm, off, 32));
  if (lane == 0) red_s[wave] = lm;
  __syncthreads();
  if (t == 0) {
    float m = red_s[0];
    for (int w = 1; w < NW; ++w) m = fmaxf(m, red_s[w]);
    red_s[0] = m;
  }
  __syncthreads();
  const float smax = red_s[0];
  __syncthreads();

  float ls = 0.0f;
  for (int l = t; l < L; l += BDIM) {
    const float e = __expf(s_s[l] - smax);
    s_s[l] = e;
    ls += e;
  }
#pragma unroll
  for (int off = 16; off > 0; off >>= 1) ls += __shfl_xor(ls, off, 32);
  if (lane == 0) red_s[wave] = ls;
  __syncthreads();
  if (t == 0) {
    float s = 0.0f;
    for (int w = 0; w < NW; ++w) s += red_s[w];
    red_s[0] = s;
  }
  __syncthreads();
  const float inv = 1.0f / red_s[0];

  // context: thread t owns column d=t (coalesced over d for each l)
  float acc = 0.0f;
#pragma unroll 4
  for (int l = 0; l < L; ++l)
    acc += s_s[l] * v[((size_t)b * L + l) * D + t];
  Cout[(size_t)bi * D + t] = acc * inv;
}

// ---------------------------------------------------------------------------
// W_hh [3H,H] -> W_hhT [H,3H]
// ---------------------------------------------------------------------------
__global__ void transpose_whh(const float* __restrict__ W_hh,
                              float* __restrict__ Wt) {
  using namespace sm;
  const int idx = blockIdx.x * 256 + threadIdx.x;
  if (idx < G3 * H) {
    const int row = idx / H, col = idx % H;
    Wt[(size_t)col * G3 + row] = W_hh[idx];
  }
}

// ---------------------------------------------------------------------------
// GRU recurrence: block per batch b (48 independent chains), 384 threads.
// ---------------------------------------------------------------------------
__global__ __launch_bounds__(384)
void gru_kernel(const float* __restrict__ GI,
                const float* __restrict__ WhhT,   // [H, 3H]
                const float* __restrict__ b_hh,
                float* __restrict__ hs,           // [B,L,H]
                float* __restrict__ h_last) {     // [B,H]
  using namespace sm;
  const int b = blockIdx.x;
  const int j = threadIdx.x;

  __shared__ float h_s[H];
  h_s[j] = 0.0f;
  __syncthreads();

  const float br = b_hh[j], bz = b_hh[j + H], bn = b_hh[j + 2 * H];

  for (int i = 0; i < L; ++i) {
    float gr = br, gz = bz, gn = bn;
#pragma unroll 4
    for (int k = 0; k < H; ++k) {
      const float hk = h_s[k];                    // LDS broadcast
      const float* wr = WhhT + (size_t)k * G3 + j;
      gr += wr[0]     * hk;
      gz += wr[H]     * hk;
      gn += wr[2 * H] * hk;
    }
    const float* gi = GI + ((size_t)b * L + i) * G3;
    const float ir  = gi[j], iz = gi[j + H], inn = gi[j + 2 * H];
    const float rg = 1.0f / (1.0f + __expf(-(ir + gr)));
    const float zg = 1.0f / (1.0f + __expf(-(iz + gz)));
    const float ng = tanhf(inn + rg * gn);
    const float hnew = (1.0f - zg) * ng + zg * h_s[j];
    __syncthreads();
    h_s[j] = hnew;
    hs[((size_t)b * L + i) * H + j] = hnew;
    __syncthreads();
  }
  h_last[(size_t)b * H + j] = h_s[j];
}

// ---------------------------------------------------------------------------
extern "C" void kernel_launch(void* const* d_in, const int* in_sizes, int n_in,
                              void* d_out, int out_size, void* d_ws, size_t ws_size,
                              hipStream_t stream) {
  using namespace sm;
  (void)in_sizes; (void)n_in; (void)out_size; (void)ws_size;

  const float*         v      = (const float*)d_in[0];
  const unsigned char* mask   = (const unsigned char*)d_in[1];   // jnp.bool_
  const float*         Wp_cur = (const float*)d_in[2];
  const float*         Wp_seq = (const float*)d_in[3];
  const float*         Vw     = (const float*)d_in[4];
  const float*         W_ih   = (const float*)d_in[5];           // [3H, 2D]
  const float*         W_hh   = (const float*)d_in[6];           // [3H, H]
  const float*         b_ih   = (const float*)d_in[7];
  const float*         b_hh   = (const float*)d_in[8];

  float* hs     = (float*)d_out;                 // [B,L,H]
  float* h_last = hs + (size_t)M * H;            // [B,H]

  // Workspace layout (f32): ~218 MB total.
  char* ws = (char*)d_ws;
  float* Wpv  = (float*)(ws);                                     // [M,H]
  float* Wpi  = (float*)(ws + (size_t)M * H * 4);                 // [M,H]
  float* Cctx = (float*)(ws + (size_t)2 * M * H * 4);             // [M,D]
  float* GI   = (float*)(ws + (size_t)3 * M * H * 4);             // [M,3H]
  float* WhhT = (float*)(ws + (size_t)3 * M * H * 4
                            + (size_t)M * G3 * 4);                // [H,3H]

  const dim3 gblk(32, 8);   // 8 waves/block, 128x64 tile per block

  // 1,2) projections: [M,384] = v[M,384] @ W[384,384]
  wmma_gemm<false, false, false><<<dim3(H / 64, M / 128), gblk, 0, stream>>>(
      v, nullptr, Wp_seq, nullptr, Wpv, D, H, 0);
  wmma_gemm<false, false, false><<<dim3(H / 64, M / 128), gblk, 0, stream>>>(
      v, nullptr, Wp_cur, nullptr, Wpi, D, H, 0);

  // 3) fused attention -> context C
  attention_kernel<<<dim3(M), dim3(384), 0, stream>>>(Wpv, Wpi, Vw, v, mask, Cctx);

  // 4) GI = [v | C] @ W_ih^T + b_ih : [M,1152]
  wmma_gemm<true, true, true><<<dim3(G3 / 64, M / 128), gblk, 0, stream>>>(
      v, Cctx, W_ih, b_ih, GI, 2 * D, G3, D);

  // 5) recurrence
  transpose_whh<<<dim3((G3 * H + 255) / 256), dim3(256), 0, stream>>>(W_hh, WhhT);
  gru_kernel<<<dim3(B), dim3(384), 0, stream>>>(GI, WhhT, b_hh, hs, h_last);
}
